// GNNModelRegression_25761213841427
// MI455X (gfx1250) — compile-verified, matrix operand order checked by reading, not run on hardware
//
#include <hip/hip_runtime.h>
#include <math.h>

#define TPB 256

typedef __attribute__((ext_vector_type(16))) _Float16 v16h;
typedef __attribute__((ext_vector_type(8)))  float    v8f;

// K-index held by (lane, element i) for the 16-bit 16x32 A-matrix WMMA layout
// (lanes 0-15: K 0..7 in v0-3, 16..23 in v4-7; lanes 16-31: +8). B mirrors A
// with M<->N (lane holds one column of B). Adjacent element pairs are adjacent
// K, so per-lane fragment reads merge into ds_load_b128.
__device__ __forceinline__ int wmma_kidx(int lane, int i) {
  int kbase = (lane < 16) ? 0 : 8;
  int p = i >> 1, sub = i & 1;
  return (p < 4) ? (kbase + 2 * p + sub) : (16 + kbase + 2 * (p - 4) + sub);
}

// One block = 8 waves = 128 rows. Weights (4x, stored N-major so B-fragment
// reads are contiguous like A) and the x-tile are staged in LDS as zero-padded
// f16 images with branch-free clamped loads. 32-bit index math throughout;
// store epilogue has a wave-uniform full-tile fast path.
template <int IN_DIM, int OUT_DIM>
__global__ void __launch_bounds__(TPB)
linear4_kernel(const float* __restrict__ x, int n_nodes,
               const float* __restrict__ Wq, const float* __restrict__ bq,
               const float* __restrict__ Wk, const float* __restrict__ bk,
               const float* __restrict__ Wv, const float* __restrict__ bv,
               const float* __restrict__ Ws, const float* __restrict__ bs,
               float* __restrict__ q, float* __restrict__ k,
               float* __restrict__ v, float* __restrict__ sk) {
  __shared__ _Float16 wlds[4][32][32];   // [matrix][N][K]  (N-major!), zero padded
  __shared__ float    blds[4][32];       // bias, zero padded
  __shared__ _Float16 xlds[8][16][32];   // [wave][row][K], zero padded

  const int tid = threadIdx.x;
  const float* Wm[4] = {Wq, Wk, Wv, Ws};
  const float* Bm[4] = {bq, bk, bv, bs};
  float*       Om[4] = {q, k, v, sk};

  // ---- stage weights N-major (branch-free: clamped address + select) ----
  for (int idx = tid; idx < 4 * 32 * 32; idx += TPB) {
    int w = idx >> 10, n = (idx >> 5) & 31, kk = idx & 31;
    int kc = kk < IN_DIM ? kk : IN_DIM - 1;
    int nc = n < OUT_DIM ? n : OUT_DIM - 1;
    float val = Wm[w][kc * OUT_DIM + nc];
    val = (kk < IN_DIM && n < OUT_DIM) ? val : 0.0f;
    wlds[w][n][kk] = (_Float16)val;
  }
  for (int idx = tid; idx < 4 * 32; idx += TPB) {
    int w = idx >> 5, n = idx & 31;
    float val = Bm[w][n < OUT_DIM ? n : OUT_DIM - 1];
    blds[w][n] = (n < OUT_DIM) ? val : 0.0f;
  }

  // ---- stage x tile: 128 rows x 32 (K-padded), 32-bit addressing ----
  const int row0b = blockIdx.x * 128;
  for (int idx = tid; idx < 128 * 32; idx += TPB) {
    int r = idx >> 5, cc = idx & 31;
    int row = row0b + r;
    int rc = row < n_nodes ? row : n_nodes - 1;
    int cclamp = cc < IN_DIM ? cc : IN_DIM - 1;
    float val = x[rc * IN_DIM + cclamp];
    val = (row < n_nodes && cc < IN_DIM) ? val : 0.0f;
    xlds[r >> 4][r & 15][cc] = (_Float16)val;
  }
  __syncthreads();

  const int waveId = tid >> 5;
  const int lane   = tid & 31;
  const int row0   = row0b + waveId * 16;
  if (row0 >= n_nodes) return;  // uniform per wave -> EXEC all-ones for WMMA

  const int m = lane & 15;
  const int mrow = (lane < 16) ? 0 : 8;  // row offset of this lane-half's D rows
  const bool fullTile = (row0 + 16 <= n_nodes);  // wave-uniform

  v16h a;
#pragma unroll
  for (int i = 0; i < 16; ++i) a[i] = xlds[waveId][m][wmma_kidx(lane, i)];

#pragma unroll
  for (int w = 0; w < 4; ++w) {
    float* out = Om[w];
#pragma unroll
    for (int nt = 0; nt < OUT_DIM; nt += 16) {
      int n = nt + m;
      v16h bmat;
#pragma unroll
      for (int i = 0; i < 16; ++i) bmat[i] = wlds[w][n][wmma_kidx(lane, i)];
      v8f c = {};
      c = __builtin_amdgcn_wmma_f32_16x16x32_f16(false, a, false, bmat,
                                                 (short)0, c, false, false);
      float bias = blds[w][n];
      int base = (row0 + mrow) * OUT_DIM + n;  // fits in 32 bits
      if (fullTile) {
#pragma unroll
        for (int r = 0; r < 8; ++r) out[base + r * OUT_DIM] = c[r] + bias;
      } else {
#pragma unroll
        for (int r = 0; r < 8; ++r) {
          if (row0 + mrow + r < n_nodes) out[base + r * OUT_DIM] = c[r] + bias;
        }
      }
    }
  }
}

__global__ void fill_kernel(float* __restrict__ p, float val, long long n) {
  long long i = (long long)blockIdx.x * blockDim.x + threadIdx.x;
  long long stride = (long long)gridDim.x * blockDim.x;
  for (; i < n; i += stride) p[i] = val;
}

__device__ __forceinline__ void atomic_max_float(float* addr, float val) {
  if (val >= 0.0f) atomicMax((int*)addr, __float_as_int(val));
  else             atomicMin((unsigned int*)addr, __float_as_uint(val));
}

template <int DIM>
__global__ void edge_score_kernel(const int* __restrict__ src, const int* __restrict__ dst,
                                  const float* __restrict__ q, const float* __restrict__ k,
                                  float* __restrict__ s_edge, float* __restrict__ smax,
                                  int n_edges, float inv_sqrt_d) {
  long long stride = (long long)gridDim.x * blockDim.x;
  for (long long e = (long long)blockIdx.x * blockDim.x + threadIdx.x; e < n_edges; e += stride) {
    int s = src[e], d = dst[e];
    const float* qd = q + (size_t)d * DIM;
    const float* ks = k + (size_t)s * DIM;
    float acc = 0.0f;
#pragma unroll
    for (int c = 0; c < DIM; ++c) acc += qd[c] * ks[c];
    acc *= inv_sqrt_d;
    s_edge[e] = acc;
    atomic_max_float(&smax[d], acc);
  }
}

__global__ void edge_exp_kernel(const int* __restrict__ dst, const float* __restrict__ smax,
                                float* __restrict__ s_edge, float* __restrict__ denom,
                                int n_edges) {
  long long stride = (long long)gridDim.x * blockDim.x;
  for (long long e = (long long)blockIdx.x * blockDim.x + threadIdx.x; e < n_edges; e += stride) {
    int d = dst[e];
    float ev = __expf(s_edge[e] - smax[d]);
    s_edge[e] = ev;  // overwrite score with weight
    atomicAdd(&denom[d], ev);
  }
}

// One thread per (edge, channel): agg[dst] += e_w * v[src]
__global__ void edge_agg_kernel(const int* __restrict__ src, const int* __restrict__ dst,
                                const float* __restrict__ e_w, const float* __restrict__ v,
                                float* __restrict__ agg, long long total, int shift) {
  long long stride = (long long)gridDim.x * blockDim.x;
  int mask = (1 << shift) - 1;
  for (long long i = (long long)blockIdx.x * blockDim.x + threadIdx.x; i < total; i += stride) {
    long long e = i >> shift;
    int c = (int)(i & mask);
    int s = src[e], d = dst[e];
    atomicAdd(&agg[((size_t)d << shift) + c], e_w[e] * v[((size_t)s << shift) + c]);
  }
}

// h = relu(agg/max(denom,1e-16) + skip), per element
__global__ void node_out_kernel(const float* __restrict__ agg, const float* __restrict__ denom,
                                const float* __restrict__ sk, float* __restrict__ h,
                                long long total, int shift) {
  long long stride = (long long)gridDim.x * blockDim.x;
  for (long long i = (long long)blockIdx.x * blockDim.x + threadIdx.x; i < total; i += stride) {
    long long n = i >> shift;
    float den = fmaxf(denom[n], 1e-16f);
    float val = agg[i] / den + sk[i];
    h[i] = fmaxf(val, 0.0f);
  }
}

__global__ void pool_kernel(const float* __restrict__ h, const int* __restrict__ batch,
                            float* __restrict__ sums, float* __restrict__ cnts, int n_nodes) {
  long long stride = (long long)gridDim.x * blockDim.x;
  for (long long n = (long long)blockIdx.x * blockDim.x + threadIdx.x; n < n_nodes; n += stride) {
    int b = batch[n];
    atomicAdd(&cnts[b], 1.0f);
    const float* hp = h + (size_t)n * 32;
    float* sp = sums + (size_t)b * 32;
#pragma unroll
    for (int c = 0; c < 32; ++c) atomicAdd(&sp[c], hp[c]);
  }
}

// g = sums/max(cnt,1); out = relu(g@Wf1+bf1)@Wf2 + bf2  (one thread per graph)
__global__ void mlp_kernel(const float* __restrict__ sums, const float* __restrict__ cnts,
                           const float* __restrict__ Wf1, const float* __restrict__ bf1,
                           const float* __restrict__ Wf2, const float* __restrict__ bf2,
                           float* __restrict__ out, int n_graphs) {
  int g = blockIdx.x * blockDim.x + threadIdx.x;
  if (g >= n_graphs) return;
  float inv = 1.0f / fmaxf(cnts[g], 1.0f);
  float gv[32];
#pragma unroll
  for (int c = 0; c < 32; ++c) gv[c] = sums[(size_t)g * 32 + c] * inv;
  float acc = bf2[0];
  for (int j = 0; j < 64; ++j) {
    float hj = bf1[j];
#pragma unroll
    for (int c = 0; c < 32; ++c) hj += gv[c] * Wf1[c * 64 + j];
    hj = fmaxf(hj, 0.0f);
    acc += hj * Wf2[j];
  }
  out[g] = acc;
}

static inline unsigned cdivu(long long a, long long b) { return (unsigned)((a + b - 1) / b); }

extern "C" void kernel_launch(void* const* d_in, const int* in_sizes, int n_in,
                              void* d_out, int out_size, void* d_ws, size_t ws_size,
                              hipStream_t stream) {
  const float* x     = (const float*)d_in[0];
  const int*   ei    = (const int*)d_in[1];
  const int*   batch = (const int*)d_in[2];
  const float* Wq1 = (const float*)d_in[3],  *bq1 = (const float*)d_in[4];
  const float* Wk1 = (const float*)d_in[5],  *bk1 = (const float*)d_in[6];
  const float* Wv1 = (const float*)d_in[7],  *bv1 = (const float*)d_in[8];
  const float* Ws1 = (const float*)d_in[9],  *bs1 = (const float*)d_in[10];
  const float* Wq2 = (const float*)d_in[11], *bq2 = (const float*)d_in[12];
  const float* Wk2 = (const float*)d_in[13], *bk2 = (const float*)d_in[14];
  const float* Wv2 = (const float*)d_in[15], *bv2 = (const float*)d_in[16];
  const float* Ws2 = (const float*)d_in[17], *bs2 = (const float*)d_in[18];
  const float* Wf1 = (const float*)d_in[19], *bf1 = (const float*)d_in[20];
  const float* Wf2 = (const float*)d_in[21], *bf2 = (const float*)d_in[22];

  const int N = in_sizes[0] / 11;
  const int E = in_sizes[1] / 2;
  const int G = out_size;
  const int* src = ei;
  const int* dst = ei + E;

  // Workspace layout (floats); hot arrays sized to fit in the 192MB L2
  float* ws = (float*)d_ws;
  const size_t ND = (size_t)N * 32;
  float* q     = ws;
  float* k     = ws + ND;
  float* v     = ws + 2 * ND;
  float* sk    = ws + 3 * ND;
  float* agg   = ws + 4 * ND;
  float* h1    = ws + 5 * ND;
  float* h2    = ws + 6 * ND;
  float* sE    = ws + 7 * ND;            // E floats (score -> edge weight)
  float* smax  = sE + (size_t)E;         // N
  float* denom = smax + N;               // N
  float* sums  = denom + N;              // G*32
  float* cnts  = sums + (size_t)G * 32;  // G

  const unsigned linBlocks = cdivu(N, 128);

  // ---------------- Layer 1: in=11, d=16 ----------------
  fill_kernel<<<cdivu(N, TPB), TPB, 0, stream>>>(smax, -INFINITY, N);
  fill_kernel<<<cdivu(N, TPB), TPB, 0, stream>>>(denom, 0.0f, N);
  fill_kernel<<<cdivu((long long)N * 16, TPB), TPB, 0, stream>>>(agg, 0.0f, (long long)N * 16);
  linear4_kernel<11, 16><<<linBlocks, TPB, 0, stream>>>(x, N,
                                                        Wq1, bq1, Wk1, bk1, Wv1, bv1, Ws1, bs1,
                                                        q, k, v, sk);
  edge_score_kernel<16><<<cdivu(E, TPB), TPB, 0, stream>>>(src, dst, q, k, sE, smax, E, 0.25f);
  edge_exp_kernel<<<cdivu(E, TPB), TPB, 0, stream>>>(dst, smax, sE, denom, E);
  edge_agg_kernel<<<cdivu((long long)E * 16, TPB), TPB, 0, stream>>>(src, dst, sE, v, agg,
                                                                     (long long)E * 16, 4);
  node_out_kernel<<<cdivu((long long)N * 16, TPB), TPB, 0, stream>>>(agg, denom, sk, h1,
                                                                     (long long)N * 16, 4);

  // ---------------- Layer 2: in=16, d=32 ----------------
  fill_kernel<<<cdivu(N, TPB), TPB, 0, stream>>>(smax, -INFINITY, N);
  fill_kernel<<<cdivu(N, TPB), TPB, 0, stream>>>(denom, 0.0f, N);
  fill_kernel<<<cdivu((long long)N * 32, TPB), TPB, 0, stream>>>(agg, 0.0f, (long long)N * 32);
  linear4_kernel<16, 32><<<linBlocks, TPB, 0, stream>>>(h1, N,
                                                        Wq2, bq2, Wk2, bk2, Wv2, bv2, Ws2, bs2,
                                                        q, k, v, sk);
  edge_score_kernel<32><<<cdivu(E, TPB), TPB, 0, stream>>>(src, dst, q, k, sE, smax, E,
                                                           0.17677669529663687f);
  edge_exp_kernel<<<cdivu(E, TPB), TPB, 0, stream>>>(dst, smax, sE, denom, E);
  edge_agg_kernel<<<cdivu((long long)E * 32, TPB), TPB, 0, stream>>>(src, dst, sE, v, agg,
                                                                     (long long)E * 32, 5);
  node_out_kernel<<<cdivu((long long)N * 32, TPB), TPB, 0, stream>>>(agg, denom, sk, h2,
                                                                     (long long)N * 32, 5);

  // ---------------- Pool + MLP ----------------
  fill_kernel<<<cdivu((long long)G * 32, TPB), TPB, 0, stream>>>(sums, 0.0f, (long long)G * 32);
  fill_kernel<<<cdivu(G, TPB), TPB, 0, stream>>>(cnts, 0.0f, G);
  pool_kernel<<<cdivu(N, TPB), TPB, 0, stream>>>(h2, batch, sums, cnts, N);
  mlp_kernel<<<cdivu(G, TPB), TPB, 0, stream>>>(sums, cnts, Wf1, bf1, Wf2, bf2,
                                                (float*)d_out, G);
}
